// ConversationalMoE_996432413527
// MI455X (gfx1250) — matrix operand order repaired and checked by reading, not compiled, not run on hardware
//
#include <hip/hip_runtime.h>
#include <hip/hip_bf16.h>

// Problem dims (fixed by the reference)
#define Bb  4
#define Ss  1024
#define Vv  32000
#define Hh  1024
#define Ee  8
#define NHh 4
#define HDd 256
#define Ii  2048
#define CHh 256
#define Nn  (Bb*Ss)
#define BNH (Bb*NHh)

// Use the gfx1250 async global->LDS path when the toolchain exposes it.
#if defined(__has_builtin)
#if __has_builtin(__builtin_amdgcn_global_load_async_to_lds_b128) && \
    __has_builtin(__builtin_amdgcn_s_wait_asynccnt)
#define CMOE_ASYNC_LDS 1
#else
#define CMOE_ASYNC_LDS 0
#endif
#else
#define CMOE_ASYNC_LDS 0
#endif

typedef __attribute__((ext_vector_type(16))) __bf16 v16bf;
typedef __attribute__((ext_vector_type(8)))  float  v8f;

// GCC-style vector to match the async builtin's parameter type exactly:
//   'int __attribute__((vector_size(16))) __device__ *'
typedef int v4i_vec __attribute__((vector_size(16)));
typedef __attribute__((address_space(1))) v4i_vec as1_v4i;
typedef __attribute__((address_space(3))) v4i_vec as3_v4i;

union FragU { v16bf v; uint4 u[2]; };

__device__ __forceinline__ unsigned short f2bf(float f) {
  unsigned int u = __float_as_uint(f);
  u += 0x7FFFu + ((u >> 16) & 1u);           // round-to-nearest-even
  return (unsigned short)(u >> 16);
}

__device__ __forceinline__ float gelu_exact(float x) {
  return 0.5f * x * (1.0f + erff(x * 0.70710678118654752f));
}

// A-matrix 16x32 bf16 fragment (ISA 7.12.2): lane = M row (lane&15);
// lanes 0-15 hold K = k0+[0..7] and k0+16+[0..7]; lanes 16-31 hold +8.
__device__ __forceinline__ v16bf load_fragA(const unsigned short* base, int ld,
                                            int row, int k0, int lane) {
  const unsigned short* p = base + (size_t)row * ld + k0 + ((lane & 16) ? 8 : 0);
  FragU f;
  f.u[0] = *(const uint4*)(p);
  f.u[1] = *(const uint4*)(p + 16);
  return f.v;
}

// ---------------------------------------------------------------------------
// Batched GEMM: C[z] = epilogue( A[z] @ W[z]^T )
// Block = 256 threads = 8 waves; tile 128(M) x 64(N); K step 32.
// B tile (64 rows x 32 K) staged in LDS, double-buffered via async->LDS.
// Epilogue: r = dot + bias[col]; act; r *= mul[row,col]; r *= rowscale[row];
//           Cf (+)= r (fp32), Cb = bf16(r).
// ---------------------------------------------------------------------------
#define LDSB_STRIDE 40   // shorts per staged row: 32 data + 8 pad = 80 B (16B aligned, 20-bank stride)

__global__ void __launch_bounds__(256)
cmoe_gemm_bf16_wmma(const unsigned short* __restrict__ A, long long sAz, int lda,
                    const unsigned short* __restrict__ Bw, long long sBz, int ldb,
                    const float* __restrict__ bias,
                    float* __restrict__ Cf,
                    unsigned short* __restrict__ Cb,
                    long long sCz, int ldc,
                    const float* __restrict__ mul, int ldmul,
                    const float* __restrict__ rowscale, int rsStride,
                    int M, int Nout, int K, int act, int accumulate)
{
  __shared__ __align__(16) unsigned short ldsB[2][64 * LDSB_STRIDE];

  const int tid  = threadIdx.x;
  const int lane = tid & 31;
  const int wave = tid >> 5;
  const int z    = blockIdx.z;
  A  += (size_t)z * sAz;
  Bw += (size_t)z * sBz;

  const int mBase = blockIdx.x * 128 + wave * 16;
  const int nBase = blockIdx.y * 64;
  const int rowA  = mBase + (lane & 15);

  // Cooperative staging: 256 threads x 16 B = 64 rows x 64 B (32 bf16 each).
  const int ldRow   = tid >> 2;
  const int ldChunk = tid & 3;
  const unsigned short* gB = Bw + (size_t)(nBase + ldRow) * ldb + ldChunk * 8;
  unsigned short* lB0 = &ldsB[0][ldRow * LDSB_STRIDE + ldChunk * 8];
  unsigned short* lB1 = &ldsB[1][ldRow * LDSB_STRIDE + ldChunk * 8];

  v8f acc[4] = {};

  auto stage = [&](int buf, int k0) {
    unsigned short* dst = buf ? lB1 : lB0;
#if CMOE_ASYNC_LDS
    __builtin_amdgcn_global_load_async_to_lds_b128(
        (as1_v4i*)(gB + k0), (as3_v4i*)dst, 0, 0);
#else
    *(uint4*)dst = *(const uint4*)(gB + k0);
#endif
  };

  auto compute = [&](int buf, int k0) {
    const v16bf a = load_fragA(A, lda, rowA, k0, lane);
    const unsigned short* base = buf ? &ldsB[1][0] : &ldsB[0][0];
#pragma unroll
    for (int t = 0; t < 4; ++t) {
      // B fragment: lane = N column (lane&15); lanes 0-15 K=k0+[0..7] and k0+16+[0..7],
      // lanes 16-31 hold +8 (same order as A).
      const unsigned short* p =
          base + (size_t)(t * 16 + (lane & 15)) * LDSB_STRIDE + ((lane & 16) ? 8 : 0);
      FragU f;
      f.u[0] = *(const uint4*)(p);
      f.u[1] = *(const uint4*)(p + 16);
      acc[t] = __builtin_amdgcn_wmma_f32_16x16x32_bf16(
          false, a, false, f.v, (short)0, acc[t], false, false);
    }
  };

  const int nk = K >> 5;
  stage(0, 0);
  int i = 0;
  for (; i < nk - 1; ++i) {
    stage((i + 1) & 1, (i + 1) << 5);
    __builtin_prefetch((const void*)(A + (size_t)rowA * lda + ((i + 1) << 5)), 0, 3);
#if CMOE_ASYNC_LDS
    __builtin_amdgcn_s_wait_asynccnt(1);   // current buffer's async writes done (this wave)
#endif
    __syncthreads();                       // ... and everyone else's
    compute(i & 1, i << 5);
    __syncthreads();                       // release buffer before it is restaged
  }
#if CMOE_ASYNC_LDS
  __builtin_amdgcn_s_wait_asynccnt(0);
#endif
  __syncthreads();
  compute(i & 1, i << 5);

  // C/D layout: lanes 0-15: VGPR j -> M = mBase+j; lanes 16-31: M = mBase+j+8.
  const int rOff = (lane & 16) ? 8 : 0;
#pragma unroll
  for (int t = 0; t < 4; ++t) {
    const int col = nBase + t * 16 + (lane & 15);
    const float bv = bias ? bias[col] : 0.0f;
#pragma unroll
    for (int j = 0; j < 8; ++j) {
      const int row = mBase + rOff + j;
      float r = acc[t][j] + bv;
      if (act == 1) r = gelu_exact(r);
      if (mul)      r *= mul[(size_t)row * ldmul + col];
      if (rowscale) r *= rowscale[(size_t)row * rsStride];
      const size_t ci = (size_t)z * (size_t)sCz + (size_t)row * ldc + col;
      if (Cf) {
        float prev = accumulate ? Cf[ci] : 0.0f;
        Cf[ci] = prev + r;
      }
      if (Cb) Cb[ci] = f2bf(r);
    }
  }
}

// ---------------------------------------------------------------------------
// Elementwise / reduction helper kernels
// ---------------------------------------------------------------------------
__global__ void __launch_bounds__(256)
cmoe_convert_f32_bf16(const float* __restrict__ in, unsigned short* __restrict__ out,
                      long long n4)
{
  long long i = (long long)blockIdx.x * 256 + threadIdx.x;
  if (i >= n4) return;
  float4 v = ((const float4*)in)[i];
  ushort4 r;
  r.x = f2bf(v.x); r.y = f2bf(v.y); r.z = f2bf(v.z); r.w = f2bf(v.w);
  ((ushort4*)out)[i] = r;
}

__global__ void __launch_bounds__(256)
cmoe_embed(const int* __restrict__ ids, const float* __restrict__ tok,
           const float* __restrict__ pos, float* __restrict__ h)
{
  const int t  = blockIdx.x;          // token index in [0, N)
  const int s  = t % Ss;
  const int id = ids[t];
  float4 a = ((const float4*)(tok + (size_t)id * Hh))[threadIdx.x];
  float4 p = ((const float4*)(pos + (size_t)s  * Hh))[threadIdx.x];
  float4 r; r.x = a.x + p.x; r.y = a.y + p.y; r.z = a.z + p.z; r.w = a.w + p.w;
  ((float4*)(h + (size_t)t * Hh))[threadIdx.x] = r;
}

__global__ void __launch_bounds__(256)
cmoe_layernorm_bf16(const float* __restrict__ x, const float* __restrict__ g,
                    const float* __restrict__ beta, unsigned short* __restrict__ out)
{
  __shared__ float red[256];
  const int row = blockIdx.x;
  const float4 v = ((const float4*)(x + (size_t)row * Hh))[threadIdx.x];

  red[threadIdx.x] = v.x + v.y + v.z + v.w;
  __syncthreads();
  for (int o = 128; o > 0; o >>= 1) {
    if (threadIdx.x < o) red[threadIdx.x] += red[threadIdx.x + o];
    __syncthreads();
  }
  const float mean = red[0] * (1.0f / Hh);
  __syncthreads();

  const float dx = v.x - mean, dy = v.y - mean, dz = v.z - mean, dw = v.w - mean;
  red[threadIdx.x] = dx * dx + dy * dy + dz * dz + dw * dw;
  __syncthreads();
  for (int o = 128; o > 0; o >>= 1) {
    if (threadIdx.x < o) red[threadIdx.x] += red[threadIdx.x + o];
    __syncthreads();
  }
  const float rstd = rsqrtf(red[0] * (1.0f / Hh) + 1e-5f);

  const int i = threadIdx.x * 4;
  unsigned short* o4 = out + (size_t)row * Hh + i;
  o4[0] = f2bf(dx * rstd * g[i + 0] + beta[i + 0]);
  o4[1] = f2bf(dy * rstd * g[i + 1] + beta[i + 1]);
  o4[2] = f2bf(dz * rstd * g[i + 2] + beta[i + 2]);
  o4[3] = f2bf(dw * rstd * g[i + 3] + beta[i + 3]);
}

// rows of length S; scale 1/sqrt(HD)=1/16 applied before softmax
__global__ void __launch_bounds__(256)
cmoe_softmax_bf16(const float* __restrict__ sc, unsigned short* __restrict__ att)
{
  __shared__ float red[256];
  const int row = blockIdx.x;
  const float4 v = ((const float4*)(sc + (size_t)row * Ss))[threadIdx.x];
  const float a0 = v.x * 0.0625f, a1 = v.y * 0.0625f,
              a2 = v.z * 0.0625f, a3 = v.w * 0.0625f;

  red[threadIdx.x] = fmaxf(fmaxf(a0, a1), fmaxf(a2, a3));
  __syncthreads();
  for (int o = 128; o > 0; o >>= 1) {
    if (threadIdx.x < o) red[threadIdx.x] = fmaxf(red[threadIdx.x], red[threadIdx.x + o]);
    __syncthreads();
  }
  const float m = red[0];
  __syncthreads();

  const float e0 = __expf(a0 - m), e1 = __expf(a1 - m),
              e2 = __expf(a2 - m), e3 = __expf(a3 - m);
  red[threadIdx.x] = e0 + e1 + e2 + e3;
  __syncthreads();
  for (int o = 128; o > 0; o >>= 1) {
    if (threadIdx.x < o) red[threadIdx.x] += red[threadIdx.x + o];
    __syncthreads();
  }
  const float inv = 1.0f / red[0];

  unsigned short* o4 = att + (size_t)row * Ss + threadIdx.x * 4;
  o4[0] = f2bf(e0 * inv); o4[1] = f2bf(e1 * inv);
  o4[2] = f2bf(e2 * inv); o4[3] = f2bf(e3 * inv);
}

// [B,S,NH*HD] bf16 -> [B*NH, S, HD] bf16
__global__ void __launch_bounds__(256)
cmoe_split_heads(const unsigned short* __restrict__ in, unsigned short* __restrict__ out)
{
  const int idx = blockIdx.x * 256 + threadIdx.x;
  const int d  = idx % HDd;
  const int s  = (idx / HDd) % Ss;
  const int zz = idx / (HDd * Ss);
  const int b  = zz / NHh, hh = zz % NHh;
  out[idx] = in[((size_t)(b * Ss + s)) * Hh + hh * HDd + d];
}

// [B,S,NH*HD] bf16 -> [B*NH, HD, S] bf16 (transposed V for att@V as A@W^T)
__global__ void __launch_bounds__(256)
cmoe_transpose_v(const unsigned short* __restrict__ in, unsigned short* __restrict__ out)
{
  const int idx = blockIdx.x * 256 + threadIdx.x;
  const int s  = idx % Ss;
  const int d  = (idx / Ss) % HDd;
  const int zz = idx / (Ss * HDd);
  const int b  = zz / NHh, hh = zz % NHh;
  out[idx] = in[((size_t)(b * Ss + s)) * Hh + hh * HDd + d];
}

// [B*NH, S, HD] f32 -> [N, H] bf16
__global__ void __launch_bounds__(256)
cmoe_merge_heads(const float* __restrict__ oh, unsigned short* __restrict__ out)
{
  const int idx = blockIdx.x * 256 + threadIdx.x;
  const int c = idx % Hh;
  const int t = idx / Hh;
  const int b = t / Ss, s = t % Ss;
  const int hh = c / HDd, d = c % HDd;
  out[idx] = f2bf(oh[(((size_t)(b * NHh + hh)) * Ss + s) * HDd + d]);
}

// router: logits, softmax over E, top-2, renormalize -> dense we[N,E]
__global__ void __launch_bounds__(256)
cmoe_router(const float* __restrict__ ctx, const float* __restrict__ gw,
            const float* __restrict__ temp, float* __restrict__ we)
{
  const int n = blockIdx.x * 256 + threadIdx.x;
  if (n >= Nn) return;
  const float invT = 1.0f / temp[0];
  float lg[Ee];
  const float* c = ctx + (size_t)n * CHh;
#pragma unroll
  for (int e = 0; e < Ee; ++e) {
    const float* w = gw + (size_t)e * CHh;
    float s = 0.0f;
    for (int i = 0; i < CHh; ++i) s += c[i] * w[i];
    lg[e] = s * invT;
  }
  float m = lg[0];
#pragma unroll
  for (int e = 1; e < Ee; ++e) m = fmaxf(m, lg[e]);
  float sum = 0.0f;
#pragma unroll
  for (int e = 0; e < Ee; ++e) { lg[e] = __expf(lg[e] - m); sum += lg[e]; }
  const float inv = 1.0f / sum;
#pragma unroll
  for (int e = 0; e < Ee; ++e) lg[e] *= inv;

  int i0 = 0;
#pragma unroll
  for (int e = 1; e < Ee; ++e) if (lg[e] > lg[i0]) i0 = e;
  int i1 = (i0 == 0) ? 1 : 0;
#pragma unroll
  for (int e = 0; e < Ee; ++e) if (e != i0 && lg[e] > lg[i1]) i1 = e;

  const float s2 = lg[i0] + lg[i1];
  const float w0 = lg[i0] / s2, w1 = lg[i1] / s2;
#pragma unroll
  for (int e = 0; e < Ee; ++e)
    we[(size_t)n * Ee + e] = (e == i0) ? w0 : ((e == i1) ? w1 : 0.0f);
}

// ---------------------------------------------------------------------------
// Host-side orchestration
// ---------------------------------------------------------------------------
extern "C" void kernel_launch(void* const* d_in, const int* in_sizes, int n_in,
                              void* d_out, int out_size, void* d_ws, size_t ws_size,
                              hipStream_t stream) {
  (void)in_sizes; (void)n_in; (void)out_size; (void)ws_size;

  const int*   input_ids = (const int*)  d_in[0];
  const float* tok_emb   = (const float*)d_in[1];
  const float* pos_emb   = (const float*)d_in[2];
  const float* ln1_g = (const float*)d_in[3];
  const float* ln1_b = (const float*)d_in[4];
  const float* ln2_g = (const float*)d_in[5];
  const float* ln2_b = (const float*)d_in[6];
  const float* wq = (const float*)d_in[7];   const float* bq = (const float*)d_in[8];
  const float* wk = (const float*)d_in[9];   const float* bk = (const float*)d_in[10];
  const float* wv = (const float*)d_in[11];  const float* bv = (const float*)d_in[12];
  const float* wo = (const float*)d_in[13];  const float* bo = (const float*)d_in[14];
  const float* ctx_w = (const float*)d_in[15];
  const float* ctx_b = (const float*)d_in[16];
  const float* gate_w = (const float*)d_in[17];
  const float* temp   = (const float*)d_in[18];
  const float* eg = (const float*)d_in[19];
  const float* eu = (const float*)d_in[20];
  const float* ed = (const float*)d_in[21];
  const float* out_w = (const float*)d_in[22];
  const float* out_b = (const float*)d_in[23];
  float* logits = (float*)d_out;

  // bump allocator over d_ws
  char* wsb = (char*)d_ws;
  size_t off = 0;
  auto alloc = [&](size_t bytes) -> void* {
    void* p = wsb + off;
    off = (off + bytes + 255) & ~(size_t)255;
    return p;
  };

  const size_t NH_elems = (size_t)Nn * Hh;            // 4M
  float*          h    = (float*)         alloc(NH_elems * 4);
  unsigned short* xb   = (unsigned short*)alloc(NH_elems * 2);
  unsigned short* qb   = (unsigned short*)alloc(NH_elems * 2);
  unsigned short* kb   = (unsigned short*)alloc(NH_elems * 2);
  unsigned short* vb   = (unsigned short*)alloc(NH_elems * 2);
  unsigned short* qh   = (unsigned short*)alloc(NH_elems * 2);
  unsigned short* kh   = (unsigned short*)alloc(NH_elems * 2);
  unsigned short* vt   = (unsigned short*)alloc(NH_elems * 2);
  float*          scor = (float*)         alloc((size_t)BNH * Ss * Ss * 4);
  unsigned short* att  = (unsigned short*)alloc((size_t)BNH * Ss * Ss * 2);
  float*          oh   = (float*)         alloc((size_t)BNH * Ss * HDd * 4);
  unsigned short* ob   = (unsigned short*)alloc(NH_elems * 2);
  unsigned short* x2b  = (unsigned short*)alloc(NH_elems * 2);
  float*          ctxf = (float*)         alloc((size_t)Nn * CHh * 4);
  float*          wef  = (float*)         alloc((size_t)Nn * Ee * 4);
  float*          Gf   = (float*)         alloc((size_t)Nn * Ii * 4);
  unsigned short* Hb   = (unsigned short*)alloc((size_t)Nn * Ii * 2);
  unsigned short* hb   = (unsigned short*)alloc(NH_elems * 2);
  // bf16 weight copies
  unsigned short* wqb   = (unsigned short*)alloc((size_t)Hh * Hh * 2);
  unsigned short* wkb   = (unsigned short*)alloc((size_t)Hh * Hh * 2);
  unsigned short* wvb   = (unsigned short*)alloc((size_t)Hh * Hh * 2);
  unsigned short* wob   = (unsigned short*)alloc((size_t)Hh * Hh * 2);
  unsigned short* ctxwb = (unsigned short*)alloc((size_t)CHh * Hh * 2);
  unsigned short* egb   = (unsigned short*)alloc((size_t)Ee * Ii * Hh * 2);
  unsigned short* eub   = (unsigned short*)alloc((size_t)Ee * Ii * Hh * 2);
  unsigned short* edb   = (unsigned short*)alloc((size_t)Ee * Hh * Ii * 2);
  unsigned short* outwb = (unsigned short*)alloc((size_t)Vv * Hh * 2);

  auto conv = [&](const float* src, unsigned short* dst, long long n) {
    long long n4 = n / 4;
    cmoe_convert_f32_bf16<<<dim3((unsigned)((n4 + 255) / 256)), dim3(256), 0, stream>>>(src, dst, n4);
  };
  conv(wq, wqb, (long long)Hh * Hh);
  conv(wk, wkb, (long long)Hh * Hh);
  conv(wv, wvb, (long long)Hh * Hh);
  conv(wo, wob, (long long)Hh * Hh);
  conv(ctx_w, ctxwb, (long long)CHh * Hh);
  conv(eg, egb, (long long)Ee * Ii * Hh);
  conv(eu, eub, (long long)Ee * Ii * Hh);
  conv(ed, edb, (long long)Ee * Hh * Ii);
  conv(out_w, outwb, (long long)Vv * Hh);

  // embeddings -> h (fp32 residual stream)
  cmoe_embed<<<dim3(Nn), dim3(256), 0, stream>>>(input_ids, tok_emb, pos_emb, h);

  // LN1 -> xb (bf16)
  cmoe_layernorm_bf16<<<dim3(Nn), dim3(256), 0, stream>>>(h, ln1_g, ln1_b, xb);

  // GEMM launch helper
  auto gemm = [&](const unsigned short* A, long long sAz, int lda,
                  const unsigned short* Bw, long long sBz, int ldb,
                  const float* bias, float* Cf, unsigned short* Cb,
                  long long sCz, int ldc,
                  const float* mul, int ldmul,
                  const float* rowscale, int rsStride,
                  int M, int Nout, int K, int act, int accum, int batch) {
    dim3 g((unsigned)(M / 128), (unsigned)(Nout / 64), (unsigned)batch);
    cmoe_gemm_bf16_wmma<<<g, dim3(256), 0, stream>>>(
        A, sAz, lda, Bw, sBz, ldb, bias, Cf, Cb, sCz, ldc,
        mul, ldmul, rowscale, rsStride, M, Nout, K, act, accum);
  };

  // QKV projections: [N,H] x [H,H]^T -> bf16
  gemm(xb, 0, Hh, wqb, 0, Hh, bq, nullptr, qb, 0, Hh, nullptr, 0, nullptr, 0, Nn, Hh, Hh, 0, 0, 1);
  gemm(xb, 0, Hh, wkb, 0, Hh, bk, nullptr, kb, 0, Hh, nullptr, 0, nullptr, 0, Nn, Hh, Hh, 0, 0, 1);
  gemm(xb, 0, Hh, wvb, 0, Hh, bv, nullptr, vb, 0, Hh, nullptr, 0, nullptr, 0, Nn, Hh, Hh, 0, 0, 1);

  // head reshapes
  {
    unsigned total_blocks = (unsigned)(NH_elems / 256);
    cmoe_split_heads<<<dim3(total_blocks), dim3(256), 0, stream>>>(qb, qh);
    cmoe_split_heads<<<dim3(total_blocks), dim3(256), 0, stream>>>(kb, kh);
    cmoe_transpose_v<<<dim3(total_blocks), dim3(256), 0, stream>>>(vb, vt);
  }

  // scores[z] = q[z] @ k[z]^T   (batched over B*NH)
  gemm(qh, (long long)Ss * HDd, HDd, kh, (long long)Ss * HDd, HDd, nullptr,
       scor, nullptr, (long long)Ss * Ss, Ss, nullptr, 0, nullptr, 0,
       Ss, Ss, HDd, 0, 0, BNH);

  // softmax (with 1/sqrt(HD) scale) -> att bf16
  cmoe_softmax_bf16<<<dim3(BNH * Ss), dim3(256), 0, stream>>>(scor, att);

  // o[z] = att[z] @ v[z]  via vt as W
  gemm(att, (long long)Ss * Ss, Ss, vt, (long long)HDd * Ss, Ss, nullptr,
       oh, nullptr, (long long)Ss * HDd, HDd, nullptr, 0, nullptr, 0,
       Ss, HDd, Ss, 0, 0, BNH);

  cmoe_merge_heads<<<dim3((unsigned)(NH_elems / 256)), dim3(256), 0, stream>>>(oh, ob);

  // h += o @ wo^T + bo  (residual accumulate)
  gemm(ob, 0, Hh, wob, 0, Hh, bo, h, nullptr, 0, Hh, nullptr, 0, nullptr, 0,
       Nn, Hh, Hh, 0, 1, 1);

  // LN2 -> x2b (bf16)
  cmoe_layernorm_bf16<<<dim3(Nn), dim3(256), 0, stream>>>(h, ln2_g, ln2_b, x2b);

  // ctx = gelu(x2 @ ctx_w^T + ctx_b)  (fp32 for router)
  gemm(x2b, 0, Hh, ctxwb, 0, Hh, ctx_b, ctxf, nullptr, 0, CHh, nullptr, 0, nullptr, 0,
       Nn, CHh, Hh, 1, 0, 1);

  // router -> dense per-token expert weights we[N,E]
  cmoe_router<<<dim3(Nn / 256), dim3(256), 0, stream>>>(ctxf, gate_w, temp, wef);

  // experts: h += we[:,e] * ((gelu(x2@eg^T) * (x2@eu^T)) @ ed^T)
  for (int e = 0; e < Ee; ++e) {
    const unsigned short* ege = egb + (size_t)e * Ii * Hh;
    const unsigned short* eue = eub + (size_t)e * Ii * Hh;
    const unsigned short* ede = edb + (size_t)e * Hh * Ii;
    // G = gelu(x2 @ eg^T)
    gemm(x2b, 0, Hh, ege, 0, Hh, nullptr, Gf, nullptr, 0, Ii, nullptr, 0, nullptr, 0,
         Nn, Ii, Hh, 1, 0, 1);
    // Hb = (x2 @ eu^T) * G   (bf16)
    gemm(x2b, 0, Hh, eue, 0, Hh, nullptr, nullptr, Hb, 0, Ii, Gf, Ii, nullptr, 0,
         Nn, Ii, Hh, 0, 0, 1);
    // h += we[:,e] * (Hb @ ed^T)
    gemm(Hb, 0, Ii, ede, 0, Ii, nullptr, h, nullptr, 0, Hh, nullptr, 0,
         wef + e, Ee, Nn, Hh, Ii, 0, 1, 1);
  }

  // logits = h @ out_w^T + out_b
  conv(h, hb, (long long)NH_elems);
  gemm(hb, 0, Hh, outwb, 0, Hh, out_b, logits, nullptr, 0, Vv, nullptr, 0, nullptr, 0,
       Nn, Vv, Hh, 0, 0, 1);
}
